// GNN_MLP_60902636257635
// MI455X (gfx1250) — matrix operand order repaired and checked by reading, not compiled, not run on hardware
//
#include <hip/hip_runtime.h>
#include <hip/hip_bf16.h>

// ---------------- problem constants ----------------
#define NN        16384      // N = K*B nodes
#define SEQ       96
#define HFEAT     100
#define OUTF      128
#define HEADS     4
#define KNODES    64
#define BGRAPH    256
#define DEG       8
#define OUTSZ     24

typedef _Float16 half8  __attribute__((ext_vector_type(8)));
typedef _Float16 half16 __attribute__((ext_vector_type(16)));
typedef float    v8f    __attribute__((ext_vector_type(8)));

// ---------------- WMMA fragment loaders (wave32, contiguous k = 16*half + e in both A and B) ----------------
// A (16x32 f16): lane L -> row = L%16, half = L/16.
// v16h elems 0..7  = A[row, k0 + half*16 + e]
// v16h elems 8..15 = A[row, k0 + half*16 + 8 + e]
static __device__ __forceinline__ half16 load_a_frag(const _Float16* __restrict__ p, int hsel) {
    half8 lo = *(const half8*)(p + hsel * 16);
    half8 hi = *(const half8*)(p + hsel * 16 + 8);
    half16 r;
#pragma unroll
    for (int i = 0; i < 8; ++i) { r[i] = lo[i]; r[8 + i] = hi[i]; }
    return r;
}
// B (32x16 f16) from column-major Bt[col][Kd]: lane L -> col = L%16, kbase = (L/16)*16,
// elems j = B[k0+kbase+j, col] -> contiguous in Bt.
static __device__ __forceinline__ half16 load_b_frag(const _Float16* __restrict__ p) {
    half8 lo = *(const half8*)(p);
    half8 hi = *(const half8*)(p + 8);
    half16 r;
#pragma unroll
    for (int i = 0; i < 8; ++i) { r[i] = lo[i]; r[8 + i] = hi[i]; }
    return r;
}

// ------------- N-blocked f16 WMMA GEMM: C[M,Nc] = A[M,Kd] * Bt[Nc,Kd]^T -------------
// Each wave computes a 16 x (16*NB) strip: A fragment loaded once per k-step,
// NB independent WMMAs issued back-to-back for latency overlap.
template <int NB>
__global__ void gemm_f16_wmma_nb(const _Float16* __restrict__ A, int lda,
                                 const _Float16* __restrict__ Bt,
                                 float* __restrict__ C, int ldc,
                                 int M, int Nc, int Kd) {
    const int wave = (blockIdx.x * blockDim.x + threadIdx.x) >> 5;
    const int lane = threadIdx.x & 31;
    const int ntiles  = Nc >> 4;
    const int ngroups = ntiles / NB;
    const int mtiles  = M >> 4;
    if (wave >= mtiles * ngroups) return;
    const int mt = wave / ngroups, ng = wave % ngroups;
    const int r = lane & 15, hsel = lane >> 4;
    const _Float16* arow = A + (size_t)(mt * 16 + r) * lda;
    const _Float16* bcol[NB];
#pragma unroll
    for (int i = 0; i < NB; ++i)
        bcol[i] = Bt + (size_t)((ng * NB + i) * 16 + r) * Kd + hsel * 16;
    v8f acc[NB];
#pragma unroll
    for (int i = 0; i < NB; ++i) acc[i] = (v8f){};
    for (int k0 = 0; k0 < Kd; k0 += 32) {
        __builtin_prefetch(arow + k0 + 64, 0, 1);
        half16 a = load_a_frag(arow + k0, hsel);
#pragma unroll
        for (int i = 0; i < NB; ++i) {
            half16 b = load_b_frag(bcol[i] + k0);
            acc[i] = __builtin_amdgcn_wmma_f32_16x16x32_f16(false, a, false, b, (short)0, acc[i], false, false);
        }
    }
    float* crow = C + (size_t)(mt * 16 + hsel * 8) * ldc + r;
#pragma unroll
    for (int i = 0; i < NB; ++i) {
        float* cp = crow + (ng * NB + i) * 16;
#pragma unroll
        for (int v = 0; v < 8; ++v) cp[(size_t)v * ldc] = acc[i][v];
    }
}

// ------- per-k MLP layer 1: hid[k] = relu(h2t[k] @ fw1t[k]^T + fb1[k]), f16 out -------
// One wave per 16 rows covers ALL 64 output cols (NB=4): A loaded once per k-step.
__global__ void mlp1_wmma(const _Float16* __restrict__ h2t,   // [64][256][128]
                          const _Float16* __restrict__ fw1t,  // [64][64][128]
                          const float* __restrict__ fb1,      // [64][64]
                          _Float16* __restrict__ hid) {       // [64][256][64]
    const int k = blockIdx.y;
    const int mt = blockIdx.x * (blockDim.x >> 5) + (threadIdx.x >> 5); // 0..15
    const int lane = threadIdx.x & 31;
    const int r = lane & 15, hsel = lane >> 4;
    const _Float16* A  = h2t  + (size_t)k * 256 * 128;
    const _Float16* Bt = fw1t + (size_t)k * 64 * 128;
    const float*    bb = fb1  + (size_t)k * 64;
    const _Float16* arow = A + (size_t)(mt * 16 + r) * 128;
    v8f acc[4];
#pragma unroll
    for (int i = 0; i < 4; ++i) acc[i] = (v8f){};
#pragma unroll
    for (int k0 = 0; k0 < 128; k0 += 32) {
        half16 a = load_a_frag(arow + k0, hsel);
#pragma unroll
        for (int i = 0; i < 4; ++i) {
            half16 b = load_b_frag(Bt + (size_t)(i * 16 + r) * 128 + hsel * 16 + k0);
            acc[i] = __builtin_amdgcn_wmma_f32_16x16x32_f16(false, a, false, b, (short)0, acc[i], false, false);
        }
    }
    _Float16* orow = hid + (size_t)k * 256 * 64 + (size_t)(mt * 16 + hsel * 8) * 64;
#pragma unroll
    for (int i = 0; i < 4; ++i) {
        const int n = i * 16 + r;
        const float bv = bb[n];
#pragma unroll
        for (int v = 0; v < 8; ++v) {
            float val = acc[i][v] + bv;
            val = val > 0.0f ? val : 0.0f;
            orow[(size_t)v * 64 + n] = (_Float16)val;
        }
    }
}

// ------- per-k MLP layer 2: pred[k] = hid[k] @ fw2t[k]^T + fb2[k] (cols<24) -------
// One wave per 16 rows covers both 16-col tiles (NB=2).
__global__ void mlp2_wmma(const _Float16* __restrict__ hid,   // [64][256][64]
                          const _Float16* __restrict__ fw2t,  // [64][32][64] (N padded)
                          const float* __restrict__ fb2,      // [64][24]
                          float* __restrict__ pred) {         // [64][256][24]
    const int k = blockIdx.y;
    const int mt = blockIdx.x * (blockDim.x >> 5) + (threadIdx.x >> 5); // 0..15
    const int lane = threadIdx.x & 31;
    const int r = lane & 15, hsel = lane >> 4;
    const _Float16* A  = hid  + (size_t)k * 256 * 64;
    const _Float16* Bt = fw2t + (size_t)k * 32 * 64;
    const _Float16* arow = A + (size_t)(mt * 16 + r) * 64;
    v8f acc[2];
    acc[0] = (v8f){}; acc[1] = (v8f){};
#pragma unroll
    for (int k0 = 0; k0 < 64; k0 += 32) {
        half16 a = load_a_frag(arow + k0, hsel);
#pragma unroll
        for (int i = 0; i < 2; ++i) {
            half16 b = load_b_frag(Bt + (size_t)(i * 16 + r) * 64 + hsel * 16 + k0);
            acc[i] = __builtin_amdgcn_wmma_f32_16x16x32_f16(false, a, false, b, (short)0, acc[i], false, false);
        }
    }
    float* orow = pred + (size_t)k * 256 * OUTSZ + (size_t)(mt * 16 + hsel * 8) * OUTSZ;
#pragma unroll
    for (int i = 0; i < 2; ++i) {
        const int n = i * 16 + r;
        if (n < OUTSZ) {
            const float bv = fb2[(size_t)k * OUTSZ + n];
#pragma unroll
            for (int v = 0; v < 8; ++v) orow[(size_t)v * OUTSZ + n] = acc[i][v] + bv;
        }
    }
}

// ---------------- attention coefficients: al[n][0..3]=src dot, al[n][4..7]=dst dot ----------------
__global__ void compute_al(const float* __restrict__ xl, const float* __restrict__ asrc,
                           const float* __restrict__ adst, float* __restrict__ al, int C) {
    const int wid = (blockIdx.x * blockDim.x + threadIdx.x) >> 5;
    const int lane = threadIdx.x & 31;
    if (wid >= NN) return;
    const float* row = xl + (size_t)wid * (HEADS * C);
#pragma unroll
    for (int h = 0; h < HEADS; ++h) {
        float ss = 0.0f, sd = 0.0f;
        for (int c = lane; c < C; c += 32) {
            float v = row[h * C + c];
            ss += v * asrc[h * C + c];
            sd += v * adst[h * C + c];
        }
        for (int o = 16; o > 0; o >>= 1) {
            ss += __shfl_down(ss, o, 32);
            sd += __shfl_down(sd, o, 32);
        }
        if (lane == 0) { al[(size_t)wid * 8 + h] = ss; al[(size_t)wid * 8 + 4 + h] = sd; }
    }
}

// ---------------- 9-edge softmax + aggregation, layer 1 (elu, pad to 128, f16 out) ----------------
__global__ void agg_layer1(const float* __restrict__ xl, const float* __restrict__ al,
                           const int* __restrict__ esrc, const float* __restrict__ bias,
                           _Float16* __restrict__ h1h) {
    __shared__ int   s_src[9];
    __shared__ float s_alpha[9][4];
    __shared__ float s_w[9][4];
    const int n = blockIdx.x, tid = threadIdx.x;
    if (tid < 9) s_src[tid] = (tid < 8) ? esrc[n * DEG + tid] : n;
    __syncthreads();
    if (tid < 36) {
        int e = tid >> 2, h = tid & 3;
        float a = al[(size_t)s_src[e] * 8 + h] + al[(size_t)n * 8 + 4 + h];
        s_alpha[e][h] = a > 0.0f ? a : 0.2f * a;
    }
    __syncthreads();
    if (tid < 4) {
        int h = tid; float m = -1e30f;
        for (int e = 0; e < 9; ++e) m = fmaxf(m, s_alpha[e][h]);
        float t[9], s = 0.0f;
        for (int e = 0; e < 9; ++e) { t[e] = __expf(s_alpha[e][h] - m); s += t[e]; }
        float inv = 1.0f / (s + 1e-16f);
        for (int e = 0; e < 9; ++e) s_w[e][h] = t[e] * inv;
    }
    __syncthreads();
    for (int c = tid; c < HFEAT; c += blockDim.x) {
        float acc = 0.0f;
#pragma unroll
        for (int e = 0; e < 9; ++e) {
            const float* row = xl + (size_t)s_src[e] * (HEADS * HFEAT);
            acc += s_w[e][0] * row[c] + s_w[e][1] * row[HFEAT + c] +
                   s_w[e][2] * row[2 * HFEAT + c] + s_w[e][3] * row[3 * HFEAT + c];
        }
        float val = acc * 0.25f + bias[c];
        val = val > 0.0f ? val : (__expf(val) - 1.0f);            // ELU
        h1h[(size_t)n * 128 + c] = (_Float16)val;
    }
    if (tid < 128 - HFEAT) h1h[(size_t)n * 128 + HFEAT + tid] = (_Float16)0.0f; // K-pad
}

// ---------------- layer 2 aggregation -> f16, permuted [k][b][128] layout ----------------
__global__ void agg_layer2(const float* __restrict__ xl, const float* __restrict__ al,
                           const int* __restrict__ esrc, const float* __restrict__ bias,
                           _Float16* __restrict__ h2t) {
    __shared__ int   s_src[9];
    __shared__ float s_alpha[9][4];
    __shared__ float s_w[9][4];
    const int n = blockIdx.x, tid = threadIdx.x;
    if (tid < 9) s_src[tid] = (tid < 8) ? esrc[n * DEG + tid] : n;
    __syncthreads();
    if (tid < 36) {
        int e = tid >> 2, h = tid & 3;
        float a = al[(size_t)s_src[e] * 8 + h] + al[(size_t)n * 8 + 4 + h];
        s_alpha[e][h] = a > 0.0f ? a : 0.2f * a;
    }
    __syncthreads();
    if (tid < 4) {
        int h = tid; float m = -1e30f;
        for (int e = 0; e < 9; ++e) m = fmaxf(m, s_alpha[e][h]);
        float t[9], s = 0.0f;
        for (int e = 0; e < 9; ++e) { t[e] = __expf(s_alpha[e][h] - m); s += t[e]; }
        float inv = 1.0f / (s + 1e-16f);
        for (int e = 0; e < 9; ++e) s_w[e][h] = t[e] * inv;
    }
    __syncthreads();
    const size_t obase = ((size_t)(n & 63) * BGRAPH + (size_t)(n >> 6)) * OUTF; // [k][b][:]
    for (int c = tid; c < OUTF; c += blockDim.x) {
        float acc = 0.0f;
#pragma unroll
        for (int e = 0; e < 9; ++e) {
            const float* row = xl + (size_t)s_src[e] * (HEADS * OUTF);
            acc += s_w[e][0] * row[c] + s_w[e][1] * row[OUTF + c] +
                   s_w[e][2] * row[2 * OUTF + c] + s_w[e][3] * row[3 * OUTF + c];
        }
        h2t[obase + c] = (_Float16)(acc * 0.25f + bias[c]);
    }
}

// ---------------- prep: conversions / transposes / padding / output copy ----------------
__global__ void cvt_f32_to_f16(const float* __restrict__ s, _Float16* __restrict__ d, int n) {
    int i = blockIdx.x * blockDim.x + threadIdx.x;
    if (i < n) d[i] = (_Float16)s[i];
}
__global__ void w1_transpose(const float* __restrict__ W1, _Float16* __restrict__ W1t) {
    int i = blockIdx.x * blockDim.x + threadIdx.x;          // 400*96
    if (i >= 400 * 96) return;
    int o = i / 96, kk = i % 96;
    W1t[i] = (_Float16)W1[(size_t)kk * 400 + o];
}
__global__ void w2_transpose_pad(const float* __restrict__ W2, _Float16* __restrict__ W2t) {
    int i = blockIdx.x * blockDim.x + threadIdx.x;          // 512*128
    if (i >= 512 * 128) return;
    int o = i / 128, kk = i % 128;
    W2t[i] = (kk < HFEAT) ? (_Float16)W2[(size_t)kk * 512 + o] : (_Float16)0.0f;
}
__global__ void fw1_transpose(const float* __restrict__ fw1, _Float16* __restrict__ fw1t) {
    int i = blockIdx.x * blockDim.x + threadIdx.x;          // 64*64*128
    if (i >= 64 * 64 * 128) return;
    int k = i / (64 * 128), rem = i % (64 * 128);
    int o = rem / 128, kk = rem % 128;
    fw1t[i] = (_Float16)fw1[((size_t)k * 128 + kk) * 64 + o];
}
__global__ void fw2_transpose_pad(const float* __restrict__ fw2, _Float16* __restrict__ fw2t) {
    int i = blockIdx.x * blockDim.x + threadIdx.x;          // 64*32*64
    if (i >= 64 * 32 * 64) return;
    int k = i / (32 * 64), rem = i % (32 * 64);
    int o = rem / 64, kk = rem % 64;
    fw2t[i] = (o < OUTSZ) ? (_Float16)fw2[((size_t)k * 64 + kk) * OUTSZ + o] : (_Float16)0.0f;
}
__global__ void copy_f32(const float* __restrict__ s, float* __restrict__ d, int n) {
    int i = blockIdx.x * blockDim.x + threadIdx.x;
    if (i < n) d[i] = s[i];
}

// ---------------- launcher ----------------
extern "C" void kernel_launch(void* const* d_in, const int* in_sizes, int n_in,
                              void* d_out, int out_size, void* d_ws, size_t ws_size,
                              hipStream_t stream) {
    const float* x    = (const float*)d_in[0];
    const int*   edge = (const int*)  d_in[1];      // [2,E]; src = edge[0..E-1]
    const float* y    = (const float*)d_in[3];
    const float* W1   = (const float*)d_in[4];
    const float* as1  = (const float*)d_in[5];
    const float* ad1  = (const float*)d_in[6];
    const float* b1   = (const float*)d_in[7];
    const float* W2   = (const float*)d_in[8];
    const float* as2  = (const float*)d_in[9];
    const float* ad2  = (const float*)d_in[10];
    const float* b2   = (const float*)d_in[11];
    const float* fw1  = (const float*)d_in[12];
    const float* fb1  = (const float*)d_in[13];
    const float* fw2  = (const float*)d_in[14];
    const float* fb2  = (const float*)d_in[15];
    float* out = (float*)d_out;

    // workspace carve-up (256B aligned)
    char* ws = (char*)d_ws;
    size_t off = 0;
    auto carve = [&](size_t bytes) { char* p = ws + off; off = (off + bytes + 255) & ~(size_t)255; return p; };
    _Float16* xh   = (_Float16*)carve((size_t)NN * SEQ * 2);
    _Float16* W1t  = (_Float16*)carve((size_t)400 * 96 * 2);
    float*    xl1  = (float*)   carve((size_t)NN * 400 * 4);
    float*    al1  = (float*)   carve((size_t)NN * 8 * 4);
    _Float16* h1h  = (_Float16*)carve((size_t)NN * 128 * 2);
    _Float16* W2t  = (_Float16*)carve((size_t)512 * 128 * 2);
    float*    xl2  = (float*)   carve((size_t)NN * 512 * 4);
    float*    al2  = (float*)   carve((size_t)NN * 8 * 4);
    _Float16* h2t  = (_Float16*)carve((size_t)NN * 128 * 2);
    _Float16* fw1t = (_Float16*)carve((size_t)64 * 64 * 128 * 2);
    _Float16* hid  = (_Float16*)carve((size_t)64 * 256 * 64 * 2);
    _Float16* fw2t = (_Float16*)carve((size_t)64 * 32 * 64 * 2);
    (void)ws_size;

    auto gsz = [](int n, int b) { return (n + b - 1) / b; };

    // prep
    cvt_f32_to_f16   <<<gsz(NN * SEQ, 256), 256, 0, stream>>>(x, xh, NN * SEQ);
    w1_transpose     <<<gsz(400 * 96, 256), 256, 0, stream>>>(W1, W1t);
    w2_transpose_pad <<<gsz(512 * 128, 256), 256, 0, stream>>>(W2, W2t);
    fw1_transpose    <<<gsz(64 * 64 * 128, 256), 256, 0, stream>>>(fw1, fw1t);
    fw2_transpose_pad<<<gsz(64 * 32 * 64, 256), 256, 0, stream>>>(fw2, fw2t);
    copy_f32         <<<gsz(BGRAPH * KNODES * OUTSZ, 256), 256, 0, stream>>>(
                         y, out + (size_t)KNODES * BGRAPH * OUTSZ, BGRAPH * KNODES * OUTSZ);

    // GAT layer 1: xl1 = x @ W1   [16384,96]x[96,400]; 25 N-tiles -> NB=5
    {
        int waves = (NN / 16) * (400 / 16 / 5);
        gemm_f16_wmma_nb<5><<<gsz(waves * 32, 128), 128, 0, stream>>>(xh, SEQ, W1t, xl1, 400, NN, 400, SEQ);
    }
    compute_al<<<gsz(NN * 32, 256), 256, 0, stream>>>(xl1, as1, ad1, al1, HFEAT);
    agg_layer1<<<NN, 128, 0, stream>>>(xl1, al1, edge, b1, h1h);

    // GAT layer 2: xl2 = h1 @ W2  [16384,128(pad)]x[128,512]; 32 N-tiles -> NB=4
    {
        int waves = (NN / 16) * (512 / 16 / 4);
        gemm_f16_wmma_nb<4><<<gsz(waves * 32, 128), 128, 0, stream>>>(h1h, 128, W2t, xl2, 512, NN, 512, 128);
    }
    compute_al<<<gsz(NN * 32, 256), 256, 0, stream>>>(xl2, as2, ad2, al2, OUTF);
    agg_layer2<<<NN, 128, 0, stream>>>(xl2, al2, edge, b2, h2t);

    // per-k MLPs: 16 M-tiles per k, one wave covers all N
    mlp1_wmma<<<dim3(4, 64), 128, 0, stream>>>(h2t, fw1t, fb1, hid);
    mlp2_wmma<<<dim3(4, 64), 128, 0, stream>>>(hid, fw2t, fb2, out);
}
